// HybridTCNSNN_65214783422992
// MI455X (gfx1250) — matrix operand order repaired and checked by reading, compiled, not run on hardware
//
#include <hip/hip_runtime.h>
#include <hip/hip_bf16.h>
#include <stdint.h>

typedef unsigned short u16;
typedef unsigned int   u32;

typedef __bf16 v16bf __attribute__((ext_vector_type(16)));
typedef float  v8f   __attribute__((ext_vector_type(8)));
typedef u32    u32x4 __attribute__((ext_vector_type(4)));
typedef u32    u32x4v __attribute__((ext_vector_type(4)));
typedef int    i32x4v __attribute__((ext_vector_type(4)));
typedef int    i32x8v __attribute__((ext_vector_type(8)));

#if defined(__has_builtin)
# if __has_builtin(__builtin_amdgcn_tensor_load_to_lds)
#  define HAVE_TDM 1
# endif
#endif

// ---------- bf16 helpers (bit-level, no __bf16 arithmetic) ----------
static __device__ __forceinline__ u16 f2bf(float f) {
    union { float f; u32 u; } v; v.f = f;
    u32 u = v.u;
    u32 r = (u + 0x7FFFu + ((u >> 16) & 1u)) >> 16;
    return (u16)r;
}
static __device__ __forceinline__ float bf2f(u16 h) {
    union { u32 u; float f; } v; v.u = ((u32)h) << 16;
    return v.f;
}

// =====================================================================
// bf16 WMMA GEMM / dilated causal conv1d.
//   Y[n, o] = bias[o] + sum_k sum_c Wt[k, o, c] * X[n + (k-(K-1))*dil, c]
// X: (N, Cin) bf16 rows; rows -16..-1 in front of X MUST be zero (pad).
// Wt: (K, Cout, Cin) bf16.  Y: (N, Cout) f32.
// Block = 4 waves sharing one 16-row m-tile.  The K x 16 x Cin weight
// slab is staged into LDS by the Tensor Data Mover (TENSOR_LOAD_TO_LDS,
// TENSORcnt) when the builtin exists, else by per-lane
// GLOBAL_LOAD_ASYNC_TO_LDS_B128 (ASYNCcnt).  The 16-entry bias row is
// always staged with async-to-LDS.  Causal padding via per-lane address
// select into the zero pad rows -> EXEC all-1s at every WMMA.
// Requires Cin%32==0, Cout%16==0, N%64==0, K*16*Cin <= 12288.
// =====================================================================
__global__ __launch_bounds__(128) void k_wmma_gemm(
    const u16* __restrict__ X, const u16* __restrict__ Wt,
    const float* __restrict__ bias, float* __restrict__ Y,
    int Cin, int Cout, int N, int T, int K, int dil)
{
    __shared__ u16   lw[12288];                   // up to 3*16*256 bf16
    __shared__ float lbias[16];

    const int lane = threadIdx.x & 31;
    const int wid  = threadIdx.x >> 5;
    const int mTiles  = Cout >> 4;
    const int nGroups = N >> 6;
    const int mt = blockIdx.x % mTiles;
    const int gb = blockIdx.x / mTiles;
    const int m0 = mt << 4;
    const int lm = lane & 15;
    const int hi = lane >> 4;

    // ---- bias row (64B) -> LDS via async copy (wave 0, lanes 0..3) ----
    if (wid == 0 && bias != nullptr && lane < 4) {
        u32 laddr = (u32)(uintptr_t)(&lbias[lane * 4]);
        const float* src = bias + m0 + lane * 4;
        asm volatile("global_load_async_to_lds_b128 %0, %1, off"
                     :: "v"(laddr), "v"(src) : "memory");
    }

#if defined(HAVE_TDM)
    // ---- weight slab via Tensor Data Mover ----
    // 2D tile: dim0 = 16*Cin contiguous bf16 (one tap's 16 rows),
    //          dim1 = K taps with stride Cout*Cin elements.
    if (wid == 0) {
        const u32 lds0 = (u32)(uintptr_t)(&lw[0]);
        const unsigned long long ga =
            (unsigned long long)(uintptr_t)Wt +
            (unsigned long long)m0 * (unsigned long long)Cin * 2ull;
        const u32 td0  = (u32)(16 * Cin);         // elements (data_size units)
        const u32 strd = (u32)(Cout * Cin);       // dim0 stride, elements

        u32x4v g0;
        g0[0] = 1u;                               // count=1, user descriptor
        g0[1] = lds0;                             // lds_addr
        g0[2] = (u32)ga;                          // global_addr[31:0]
        g0[3] = (u32)((ga >> 32) & 0x01FFFFFFu) | 0x80000000u; // addr[56:32]|type=2

        i32x8v g1;
        g1[0] = 0x10000;                          // data_size=1 (2 bytes)
        g1[1] = (int)((td0 & 0xFFFFu) << 16);     // tensor_dim0[15:0]
        g1[2] = (int)(((td0 >> 16) & 0xFFFFu) | ((u32)K << 16)); // dim0 hi | dim1
        g1[3] = (int)((td0 & 0xFFFFu) << 16);     // tile_dim0
        g1[4] = K;                                // tile_dim1 (tile_dim2=0)
        g1[5] = (int)strd;                        // tensor_dim0_stride[31:0]
        g1[6] = 0;                                // stride hi, dim1_stride lo
        g1[7] = 0;
        i32x4v g2 = {0, 0, 0, 0};
        i32x4v g3 = {0, 0, 0, 0};
#if __clang_major__ >= 23
        i32x8v g4 = {0, 0, 0, 0, 0, 0, 0, 0};
        __builtin_amdgcn_tensor_load_to_lds(g0, g1, g2, g3, g4, 0);
#else
        __builtin_amdgcn_tensor_load_to_lds(g0, g1, g2, g3, 0);
#endif
    }
#else
    // ---- fallback: per-lane async copy of the weight slab ----
    {
        const int E = K * 16 * Cin;               // u16 elements, %8 == 0
        for (int e = threadIdx.x * 8; e < E; e += 128 * 8) {
            int k = e / (16 * Cin);
            int r = e % (16 * Cin);
            int m = r / Cin;
            int c = r % Cin;
            const u16* src = Wt + ((size_t)k * Cout + (m0 + m)) * (size_t)Cin + c;
            u32 laddr = (u32)(uintptr_t)(&lw[e]);
            asm volatile("global_load_async_to_lds_b128 %0, %1, off"
                         :: "v"(laddr), "v"(src) : "memory");
        }
    }
#endif

    // every wave drains its own counters, then the whole block syncs
    asm volatile("s_wait_asynccnt 0x0" ::: "memory");
#if defined(HAVE_TDM)
    if (wid == 0) {
# if __has_builtin(__builtin_amdgcn_s_wait_tensorcnt)
        __builtin_amdgcn_s_wait_tensorcnt(0);
# else
        asm volatile("s_wait_tensorcnt 0x0" ::: "memory");
# endif
    }
#endif
    __syncthreads();

    const int ng = gb * 4 + wid;
    if (ng >= nGroups) return;
    const int n0 = ng << 6;

    v8f acc[4];
    #pragma unroll
    for (int j = 0; j < 4; ++j)
        #pragma unroll
        for (int r = 0; r < 8; ++r) acc[j][r] = 0.0f;

    int ncol[4];
    #pragma unroll
    for (int j = 0; j < 4; ++j) ncol[j] = n0 + 16 * j + lm;

    for (int k = 0; k < K; ++k) {
        const int shift = (k - (K - 1)) * dil;

        // per-lane shifted row; out-of-range -> zero pad rows [-16, -1]
        int np[4];
        #pragma unroll
        for (int j = 0; j < 4; ++j) {
            int n  = ncol[j];
            int nn = n + shift;
            if (shift != 0) {                      // wave-uniform branch
                int t = n % T;
                if (t + shift < 0) nn = lm - 16;   // v_cndmask, no EXEC split
            }
            np[j] = nn;
        }

        const u16* lrow = &lw[(k * 16 + lm) * Cin];

        for (int c0 = 0; c0 < Cin; c0 += 32) {
            // A operand from LDS: 2x ds_load_b128 (pairs g=0..3 then g=4..7)
            union { v16bf v; u32x4 q[2]; } A;
            const u16* ar = lrow + c0 + (hi << 3);
            A.q[0] = *(const u32x4*)(ar);
            A.q[1] = *(const u32x4*)(ar + 16);

            if (c0 + 32 < Cin)
                __builtin_prefetch(X + (long)np[0] * Cin + c0 + 32, 0, 1);

            // batch all B loads (one clause, one wait), then 4 WMMAs
            u32x4 bq[8];
            #pragma unroll
            for (int j = 0; j < 4; ++j) {
                const u16* xp = X + (long)np[j] * Cin + c0 + (hi << 3);
                bq[2 * j]     = *(const u32x4*)(xp);
                bq[2 * j + 1] = *(const u32x4*)(xp + 16);
            }
            #pragma unroll
            for (int j = 0; j < 4; ++j) {
                union { v16bf v; u32x4 q[2]; } Bm;
                Bm.q[0] = bq[2 * j];
                Bm.q[1] = bq[2 * j + 1];
                acc[j] = __builtin_amdgcn_wmma_f32_16x16x32_bf16(
                    false, A.v, false, Bm.v, (short)0, acc[j], false, false);
            }
        }
    }

    // Epilogue: C/D 16x16 f32 layout -> VGPR r: M = r + (hi?8:0), N = lane&15
    float bs[8];
    #pragma unroll
    for (int r = 0; r < 8; ++r)
        bs[r] = bias ? lbias[r + (hi << 3)] : 0.0f;
    #pragma unroll
    for (int j = 0; j < 4; ++j) {
        #pragma unroll
        for (int r = 0; r < 8; ++r) {
            int o = m0 + r + (hi << 3);
            Y[(size_t)ncol[j] * Cout + o] = acc[j][r] + bs[r];
        }
    }
}

// ---------- zero-fill (for the causal pad rows) ----------
__global__ void k_zero(u16* __restrict__ p, int n) {
    int i = blockIdx.x * 256 + threadIdx.x;
    if (i < n) p[i] = 0;
}

// ---------- fp32 (B,T,F) -> bf16 (N, F) (identity layout convert) ----
__global__ void k_pack_x(const float* __restrict__ x, u16* __restrict__ o, int n) {
    int i = blockIdx.x * 256 + threadIdx.x;
    if (i < n) o[i] = f2bf(x[i]);
}

// ---------- weight convert: (O,I,K) f32 -> (K,Opad,I) bf16, zero-pad O ----
__global__ void k_wcvt(const float* __restrict__ src, u16* __restrict__ dst,
                       int O, int Opad, int I, int K) {
    int i = blockIdx.x * 256 + threadIdx.x;
    int total = K * Opad * I;
    if (i >= total) return;
    int k = i / (Opad * I);
    int r = i % (Opad * I);
    int o = r / I;
    int c = r % I;
    float v = (o < O) ? src[((size_t)o * I + c) * K + k] : 0.0f;
    dst[i] = f2bf(v);
}

// ---------- per-channel BN stats over N rows of (N, C) f32 ----------
__global__ __launch_bounds__(256) void k_bn_stats(
    const float* __restrict__ Y, float* __restrict__ stats,
    int C, int N, int preRelu)
{
    int c = blockIdx.x;
    float s = 0.0f, ss = 0.0f;
    for (int n = threadIdx.x; n < N; n += 256) {
        float v = Y[(size_t)n * C + c];
        if (preRelu) v = fmaxf(v, 0.0f);
        s += v; ss += v * v;
    }
    __shared__ float a1[256], a2[256];
    a1[threadIdx.x] = s; a2[threadIdx.x] = ss;
    __syncthreads();
    for (int ofs = 128; ofs > 0; ofs >>= 1) {
        if (threadIdx.x < ofs) {
            a1[threadIdx.x] += a1[threadIdx.x + ofs];
            a2[threadIdx.x] += a2[threadIdx.x + ofs];
        }
        __syncthreads();
    }
    if (threadIdx.x == 0) {
        float m   = a1[0] / (float)N;
        float var = a2[0] / (float)N - m * m;
        stats[c]     = m;
        stats[C + c] = rsqrtf(var + 1e-5f);
    }
}

// ---------- fused BN / ReLU / residual-add / bf16 convert ----------
// resMode: 0 none, 1 add f32 residual then relu, 2 add bf16 residual then relu
__global__ void k_bn_apply(
    const float* __restrict__ Y, const float* __restrict__ stats,
    const float* __restrict__ g, const float* __restrict__ be,
    const float* __restrict__ resF, const u16* __restrict__ resB,
    u16* __restrict__ Z, int C, int total,
    int preRelu, int postRelu, int resMode)
{
    int i = blockIdx.x * 256 + threadIdx.x;
    if (i >= total) return;
    int c = i % C;
    float v = Y[i];
    if (preRelu) v = fmaxf(v, 0.0f);
    v = (v - stats[c]) * stats[C + c] * g[c] + be[c];
    if (postRelu) v = fmaxf(v, 0.0f);
    if (resMode == 1)      v = fmaxf(v + resF[i], 0.0f);
    else if (resMode == 2) v = fmaxf(v + bf2f(resB[i]), 0.0f);
    Z[i] = f2bf(v);
}

// ---------- TCN time-pool: mean over T per (b, c) -> comb[:, 0:256] ----
__global__ void k_pool_tcn(const u16* __restrict__ Xf, u16* __restrict__ comb) {
    int i = blockIdx.x * 256 + threadIdx.x;
    if (i >= 64 * 256) return;
    int b = i >> 8, c = i & 255;
    float s = 0.0f;
    const u16* p = Xf + ((size_t)b * 1024) * 256 + c;
    for (int t = 0; t < 1024; ++t) s += bf2f(p[(size_t)t * 256]);
    comb[b * 512 + c] = f2bf(s * (1.0f / 1024.0f));
}

// ---------- SNN input spikes: rate-coded counts over T ----------
__global__ void k_spikes(const float* __restrict__ x, u16* __restrict__ sb) {
    int i = blockIdx.x * 256 + threadIdx.x;
    if (i >= 10 * 64 * 64) return;
    int f = i & 63;
    int b = (i >> 6) & 63;
    int s = i >> 12;
    const float* xp = x + ((size_t)b * 1024) * 64 + f;
    int cnt = 0;
    for (int t = 0; t < 1024; ++t) {
        float p = 1.0f / (1.0f + __expf(-xp[(size_t)t * 64]));
        u32 h = ((u32)s * 0x9E3779B1u) ^ ((u32)b * 0x85EBCA77u) ^
                ((u32)t * 0xC2B2AE3Du) ^ ((u32)f * 0x27D4EB2Fu);
        h ^= h >> 15; h *= 0x2C1B3C6Du;
        h ^= h >> 12; h *= 0x297A2D39u;
        h ^= h >> 15;
        float u = (float)(h >> 8) * (1.0f / 16777216.0f);
        cnt += (u < p) ? 1 : 0;
    }
    sb[(size_t)(s * 64 + b) * 64 + f] = f2bf(2.0f * (float)cnt);
}

// ---------- LIF scan over NUM_STEPS=10; optional rate output ----------
__global__ void k_lif(const float* __restrict__ cur, u16* __restrict__ spk,
                      u16* __restrict__ comb, int H, int writeRates)
{
    int i = blockIdx.x * 256 + threadIdx.x;
    if (i >= 64 * H) return;
    int b = i / H, h = i % H;
    float mem = 0.0f, acc = 0.0f;
    for (int st = 0; st < 10; ++st) {
        float c = cur[(size_t)(st * 64 + b) * H + h];
        float reset = (mem > 1.0f) ? 1.0f : 0.0f;
        mem = 0.9f * mem + c - reset * 1.0f;
        float s = ((mem - 1.0f) > 0.0f) ? 1.0f : 0.0f;
        spk[(size_t)(st * 64 + b) * H + h] = f2bf(s);
        acc += s;
    }
    if (writeRates) comb[b * 512 + 256 + h] = f2bf(acc * 0.1f);
}

// ---------- f32 -> bf16 with optional relu ----------
__global__ void k_cvt_bf16(const float* __restrict__ in, u16* __restrict__ out,
                           int n, int doRelu) {
    int i = blockIdx.x * 256 + threadIdx.x;
    if (i >= n) return;
    float v = in[i];
    if (doRelu) v = fmaxf(v, 0.0f);
    out[i] = f2bf(v);
}

// ---------- final: strip pad, add b3, write logits ----------
__global__ void k_final(const float* __restrict__ Y3, const float* __restrict__ b3,
                        float* __restrict__ out) {
    int i = blockIdx.x * 64 + threadIdx.x;
    if (i >= 640) return;
    int b = i / 10, c = i % 10;
    out[i] = Y3[b * 16 + c] + b3[c];
}

// =====================================================================
// Host-side orchestration
// =====================================================================
extern "C" void kernel_launch(void* const* d_in, const int* in_sizes, int n_in,
                              void* d_out, int out_size, void* d_ws, size_t ws_size,
                              hipStream_t stream) {
    (void)in_sizes; (void)n_in; (void)out_size; (void)ws_size;

    char* ws = (char*)d_ws;
    const int B = 64, T = 1024, F = 64, N = B * T;   // N = 65536
    const size_t PAD = 0x2000;                       // 8KB zero pad (16 rows x 256ch)

    // workspace carve (bytes); activation X pointers sit after a zeroed pad
    u16*   XA   = (u16*)(ws + 0x00000000ull + PAD);  // bf16 act ping
    u16*   XB   = (u16*)(ws + 0x02010000ull + PAD);  // bf16 act pong
    u16*   Z    = (u16*)(ws + 0x04020000ull + PAD);  // bf16 intra-block
    float* Y    = (float*)(ws + 0x06030000ull);      // 64 MB f32 GEMM out
    float* YR   = (float*)(ws + 0x0A040000ull);      // 64 MB f32 residual
    u16*   WB   = (u16*)(ws + 0x0E050000ull);        //  8 MB bf16 weights
    float* ST   = (float*)(ws + 0x0E860000ull);      // stats
    u16*   SBA  = (u16*)(ws + 0x0E870000ull);        // SNN spikes ping
    u16*   SBB  = (u16*)(ws + 0x0E8D0000ull);        // SNN spikes pong
    u16*   COMB = (u16*)(ws + 0x0E930000ull);        // (64,512) bf16
    u16*   H1B  = (u16*)(ws + 0x0E940000ull);
    u16*   H2B  = (u16*)(ws + 0x0E948000ull);

    auto fin = (+[](void* const* p, int i) { return (const float*)p[i]; });

    // zero the causal-pad rows in front of each activation buffer
    k_zero<<<16, 256, 0, stream>>>((u16*)(ws + 0x00000000ull), 4096);
    k_zero<<<16, 256, 0, stream>>>((u16*)(ws + 0x02010000ull), 4096);
    k_zero<<<16, 256, 0, stream>>>((u16*)(ws + 0x04020000ull), 4096);

    // -------- bf16 weight pool --------
    size_t woff = 0;
    auto walloc = [&](size_t n) { u16* p = WB + woff; woff += n; return p; };
    auto wcvt = [&](const float* src, u16* dst, int O, int Opad, int I, int K) {
        int total = K * Opad * I;
        k_wcvt<<<(total + 255) / 256, 256, 0, stream>>>(src, dst, O, Opad, I, K);
    };
    auto gemm = [&](const u16* Xp, const u16* Wp, const float* bias, float* Yp,
                    int Cin, int Cout, int Nn, int Tt, int K, int dil) {
        int mTiles = Cout >> 4;
        int nGroups = Nn >> 6;
        int blocks = mTiles * ((nGroups + 3) / 4);
        k_wmma_gemm<<<blocks, 128, 0, stream>>>(Xp, Wp, bias, Yp,
                                                Cin, Cout, Nn, Tt, K, dil);
    };

    // input index map (setup_inputs dict order)
    struct BlkIdx { int base, cin, cout; bool ds; };
    const BlkIdx BI[3] = { {1, 64, 64, false}, {9, 64, 128, true}, {19, 128, 256, true} };

    u16 *Wc1[3], *Wc2[3], *Wcd[3];
    for (int b = 0; b < 3; ++b) {
        int ci = BI[b].cin, co = BI[b].cout, base = BI[b].base;
        Wc1[b] = walloc((size_t)3 * co * ci);
        wcvt(fin(d_in, base + 0), Wc1[b], co, co, ci, 3);
        Wc2[b] = walloc((size_t)3 * co * co);
        wcvt(fin(d_in, base + 4), Wc2[b], co, co, co, 3);
        if (BI[b].ds) {
            Wcd[b] = walloc((size_t)co * ci);
            wcvt(fin(d_in, base + 8), Wcd[b], co, co, ci, 1);
        } else Wcd[b] = nullptr;
    }
    // SNN weights: (64,64), (128,64), (256,128)
    u16* Wsn[3];
    const int SH[4] = {64, 64, 128, 256};
    const int snwi[3] = {29, 31, 33};
    for (int l = 0; l < 3; ++l) {
        Wsn[l] = walloc((size_t)SH[l + 1] * SH[l]);
        wcvt(fin(d_in, snwi[l]), Wsn[l], SH[l + 1], SH[l + 1], SH[l], 1);
    }
    // fusion weights
    u16* WF1 = walloc(128 * 512); wcvt(fin(d_in, 35), WF1, 128, 128, 512, 1);
    u16* WF2 = walloc(64 * 128);  wcvt(fin(d_in, 39), WF2, 64, 64, 128, 1);
    u16* WF3 = walloc(16 * 64);   wcvt(fin(d_in, 41), WF3, 10, 16, 64, 1); // pad 10->16

    // -------- TCN branch --------
    {
        int n = N * F;
        k_pack_x<<<(n + 255) / 256, 256, 0, stream>>>(fin(d_in, 0), XA, n);
    }
    u16* cur = XA; u16* nxt = XB;
    int dil = 1;
    for (int b = 0; b < 3; ++b) {
        int ci = BI[b].cin, co = BI[b].cout, base = BI[b].base;
        int total = N * co;
        // conv1 -> BN -> relu -> Z
        gemm(cur, Wc1[b], fin(d_in, base + 1), Y, ci, co, N, T, 3, dil);
        k_bn_stats<<<co, 256, 0, stream>>>(Y, ST, co, N, 0);
        k_bn_apply<<<(total + 255) / 256, 256, 0, stream>>>(
            Y, ST, fin(d_in, base + 2), fin(d_in, base + 3), nullptr, nullptr,
            Z, co, total, 0, 1, 0);
        // conv2 -> BN
        gemm(Z, Wc2[b], fin(d_in, base + 5), Y, co, co, N, T, 3, dil);
        k_bn_stats<<<co, 256, 0, stream>>>(Y, ST, co, N, 0);
        // residual
        if (BI[b].ds) {
            gemm(cur, Wcd[b], fin(d_in, base + 9), YR, ci, co, N, T, 1, 1);
            k_bn_apply<<<(total + 255) / 256, 256, 0, stream>>>(
                Y, ST, fin(d_in, base + 6), fin(d_in, base + 7), YR, nullptr,
                nxt, co, total, 0, 1, 1);
        } else {
            k_bn_apply<<<(total + 255) / 256, 256, 0, stream>>>(
                Y, ST, fin(d_in, base + 6), fin(d_in, base + 7), nullptr, cur,
                nxt, co, total, 0, 1, 2);
        }
        u16* t2 = cur; cur = nxt; nxt = t2;
        dil <<= 1;
    }
    k_pool_tcn<<<(64 * 256 + 255) / 256, 256, 0, stream>>>(cur, COMB);

    // -------- SNN branch (10 steps, N=640 rows) --------
    k_spikes<<<(10 * 64 * 64 + 255) / 256, 256, 0, stream>>>(fin(d_in, 0), SBA);
    u16* sIn = SBA; u16* sOut = SBB;
    const int snbi[3] = {30, 32, 34};
    for (int l = 0; l < 3; ++l) {
        int hin = SH[l], hout = SH[l + 1];
        gemm(sIn, Wsn[l], fin(d_in, snbi[l]), Y, hin, hout, 640, 640, 1, 1);
        k_lif<<<(64 * hout + 255) / 256, 256, 0, stream>>>(
            Y, sOut, COMB, hout, (l == 2) ? 1 : 0);
        u16* t2 = sIn; sIn = sOut; sOut = t2;
    }

    // -------- fusion head (rows = 64 batch) --------
    gemm(COMB, WF1, fin(d_in, 36), Y, 512, 128, 64, 64, 1, 1);    // h1 pre-relu
    k_bn_stats<<<128, 256, 0, stream>>>(Y, ST, 128, 64, 1);       // stats of relu(h1)
    k_bn_apply<<<(64 * 128 + 255) / 256, 256, 0, stream>>>(
        Y, ST, fin(d_in, 37), fin(d_in, 38), nullptr, nullptr, H1B, 128, 64 * 128, 1, 0, 0);
    gemm(H1B, WF2, fin(d_in, 40), Y, 128, 64, 64, 64, 1, 1);      // h2 pre-relu
    k_cvt_bf16<<<(64 * 64 + 255) / 256, 256, 0, stream>>>(Y, H2B, 64 * 64, 1);
    gemm(H2B, WF3, nullptr, Y, 64, 16, 64, 64, 1, 1);             // padded logits
    k_final<<<(640 + 63) / 64, 64, 0, stream>>>(Y, fin(d_in, 42), (float*)d_out);
}
